// GAT_GRU_88003879895456
// MI455X (gfx1250) — compile-verified, matrix operand order checked by reading, not run on hardware
//
#include <hip/hip_runtime.h>
#include <hip/hip_bf16.h>
#include <math.h>

// ---------------- WMMA / TDM types (CDNA5 gfx1250, wave32) ----------------
typedef __attribute__((ext_vector_type(16))) __bf16 v16bf;
typedef __attribute__((ext_vector_type(8)))  __bf16 v8bf;
typedef __attribute__((ext_vector_type(4)))  __bf16 v4bf;
typedef __attribute__((ext_vector_type(8)))  float  v8f;
typedef __attribute__((ext_vector_type(4)))  unsigned tdm_g0_t;
typedef __attribute__((ext_vector_type(8)))  int      tdm_g1_t;
typedef __attribute__((ext_vector_type(4)))  int      tdm_g2_t;

#define GAT_H 8
#define GAT_D 8
#define HD    64      // H*D
#define FIN   128
#define HID   16
#define G3    48      // 3*HID
#define TSEQ  2048
#define NEG_SLOPE 0.2f

// ---------------- helpers ----------------
__device__ __forceinline__ unsigned fflip(float f) {
    unsigned u = __float_as_uint(f);
    return u ^ ((u >> 31) ? 0xFFFFFFFFu : 0x80000000u);
}
__device__ __forceinline__ float funflip(unsigned u) {
    unsigned v = (u & 0x80000000u) ? (u ^ 0x80000000u) : ~u;
    return __uint_as_float(v);
}
__device__ __forceinline__ float lrelu(float p) { return p > 0.f ? p : NEG_SLOPE * p; }

// =====================================================================
// K1: xp = x @ W_gat   [N,128] @ [128,64] -> [N,64]
// bf16 WMMA, f32 accumulate. A tile staged Global->LDS by the Tensor
// Data Mover (TDM); B (weights) cached in LDS transposed [col][k] so a
// lane's WMMA fragment is two contiguous 16-byte runs (ds_load_b128).
// block = 256 thr (8 waves) -> 32 rows x 64 cols, one 16x16 tile/wave.
// =====================================================================
__global__ __launch_bounds__(256) void gat_proj_wmma(
    const float* __restrict__ x, const float* __restrict__ W,
    float* __restrict__ xp)
{
    __shared__ float  Atile[32 * 36];   // f32 TDM tile, 36-DWORD padded rows
    __shared__ __bf16 Wt[HD * 136];     // [col][k] transposed W, stride 136 (16B-aligned)

    const int tid  = threadIdx.x;
    const int lane = tid & 31;
    const int wave = tid >> 5;          // 0..7
    const int rt   = wave >> 2;         // row tile 0..1
    const int ct   = wave & 3;          // col tile 0..3
    const int rowBase = blockIdx.x * 32;

    // stage W transposed (coalesced global reads, i = k*64+col)
    for (int i = tid; i < FIN * HD; i += 256) {
        int col = i & 63, k = i >> 6;
        Wt[col * 136 + k] = (__bf16)W[i];
    }

    v8f acc = {};
    const int lrow = lane & 15;
    const int kb   = (lane >> 4) * 8;   // 0 or 8 (ISA 16-bit fragment layout)
    const unsigned ldsA = (unsigned)(uintptr_t)(void*)Atile;  // LDS aperture: low 32 bits = offset

#pragma unroll
    for (int kk = 0; kk < FIN; kk += 32) {
        __syncthreads();                // prior tile fully consumed (also fences Wt on iter 0)
        if (wave == 0) {
            // ---- TDM: 2D tile 32 rows x 32 f32, row stride 128 elems,
            //      LDS padding: +4 DWORDs every 32 DWORDs -> 36-DWORD rows
            uint64_t ga = (uint64_t)(uintptr_t)(x + (size_t)rowBase * FIN + kk);
            tdm_g0_t g0;
            g0[0] = 1u;                                     // count=1 (valid descriptor)
            g0[1] = ldsA;                                   // lds_addr
            g0[2] = (unsigned)ga;                           // global_addr[31:0]
            g0[3] = (unsigned)((ga >> 32) & 0x01FFFFFFu)    // global_addr[56:32]
                  | (2u << 30);                             // type=2 ("image")
            tdm_g1_t g1;
            g1[0] = (2 << 16)        // data_size = 4 bytes
                  | (1 << 20)        // pad_enable
                  | (4 << 22)        // pad_interval: 32 DWORDs
                  | (3 << 25);       // pad_amount: 4 DWORDs
            g1[1] = 0;                                      // tensor_dim0[15:0] (=0x100000)
            g1[2] = 0x10;                                   // tensor_dim0[31:16]=0x10, dim1 lo=0
            g1[3] = (32 << 16) | 0x10;                      // tile_dim0=32, tensor_dim1 hi=0x10
            g1[4] = 32;                                     // tile_dim1=32, tile_dim2=0
            g1[5] = FIN;                                    // tensor_dim0_stride = 128
            g1[6] = 0; g1[7] = 0;
            tdm_g2_t gz = {0, 0, 0, 0};                     // groups 2/3 unused (2D tile)
            tdm_g1_t gz8 = {0, 0, 0, 0, 0, 0, 0, 0};        // extra group (clang-23 6-arg form)
            __builtin_amdgcn_tensor_load_to_lds(g0, g1, gz, gz, gz8, 0);
            __builtin_amdgcn_s_wait_tensorcnt(0);
        }
        __syncthreads();                // tile visible to all waves

        // A fragment: two 32B runs of f32 -> cvt to bf16
        const int arow = rt * 16 + lrow;
        const float4* ap0 = (const float4*)&Atile[arow * 36 + kb];
        const float4* ap1 = (const float4*)&Atile[arow * 36 + kb + 16];
        float4 f0 = ap0[0], f1 = ap0[1], f2 = ap1[0], f3 = ap1[1];
        v16bf a;
        a[0]=(__bf16)f0.x;  a[1]=(__bf16)f0.y;  a[2]=(__bf16)f0.z;  a[3]=(__bf16)f0.w;
        a[4]=(__bf16)f1.x;  a[5]=(__bf16)f1.y;  a[6]=(__bf16)f1.z;  a[7]=(__bf16)f1.w;
        a[8]=(__bf16)f2.x;  a[9]=(__bf16)f2.y;  a[10]=(__bf16)f2.z; a[11]=(__bf16)f2.w;
        a[12]=(__bf16)f3.x; a[13]=(__bf16)f3.y; a[14]=(__bf16)f3.z; a[15]=(__bf16)f3.w;

        // B fragment: two contiguous 16B bf16 runs -> ds_load_b128 pair
        const int bcol = ct * 16 + lrow;
        v8bf blo = *(const v8bf*)&Wt[bcol * 136 + kk + kb];
        v8bf bhi = *(const v8bf*)&Wt[bcol * 136 + kk + kb + 16];
        v16bf b = __builtin_shufflevector(blo, bhi,
                                          0,1,2,3,4,5,6,7,8,9,10,11,12,13,14,15);

        acc = __builtin_amdgcn_wmma_f32_16x16x32_bf16(
            false, a, false, b, (short)0, acc, false, false);
    }

    const int col  = ct * 16 + lrow;
    const int mofs = (lane >> 4) * 8;   // C/D layout: VGPR r -> M=r / r+8
#pragma unroll
    for (int r = 0; r < 8; ++r) {
        int row = rowBase + rt * 16 + r + mofs;
        xp[(size_t)row * HD + col] = acc[r];
    }
}

// =====================================================================
// K2: per-node attention logits a_src/a_dst  [N,8]
// =====================================================================
__global__ __launch_bounds__(256) void att_logits(
    const float* __restrict__ xp, const float* __restrict__ att_src,
    const float* __restrict__ att_dst, float* __restrict__ asrc,
    float* __restrict__ adst, int N)
{
    int i = blockIdx.x * blockDim.x + threadIdx.x;
    if (i >= N * GAT_H) return;
    int n = i >> 3, h = i & 7;
    const float4* xr = (const float4*)(xp + (size_t)n * HD + h * GAT_D);
    const float4* s4 = (const float4*)(att_src + h * GAT_D);
    const float4* d4 = (const float4*)(att_dst + h * GAT_D);
    float4 x0 = xr[0], x1 = xr[1];
    float4 s0 = s4[0], s1 = s4[1];
    float4 dd0 = d4[0], dd1 = d4[1];
    asrc[i] = x0.x*s0.x + x0.y*s0.y + x0.z*s0.z + x0.w*s0.w
            + x1.x*s1.x + x1.y*s1.y + x1.z*s1.z + x1.w*s1.w;
    adst[i] = x0.x*dd0.x + x0.y*dd0.y + x0.z*dd0.z + x0.w*dd0.w
            + x1.x*dd1.x + x1.y*dd1.y + x1.z*dd1.z + x1.w*dd1.w;
}

// =====================================================================
// K3: init scratch (m = flip(-inf), denom = 0, agg = 0)
// =====================================================================
__global__ __launch_bounds__(256) void init_scratch(
    unsigned* __restrict__ mu, float* __restrict__ denom,
    float* __restrict__ agg, int N)
{
    int i = blockIdx.x * blockDim.x + threadIdx.x;
    if (i < N * HD) agg[i] = 0.f;
    if (i < N * GAT_H) { mu[i] = 0x007FFFFFu; denom[i] = 0.f; }
}

// =====================================================================
// K4: segment max of leaky-relu attention logits (per dst, per head)
// =====================================================================
__global__ __launch_bounds__(256) void edge_max(
    const int* __restrict__ src, const int* __restrict__ dst,
    const float* __restrict__ asrc, const float* __restrict__ adst,
    unsigned* __restrict__ mu, int E, int Etot)
{
    int e = blockIdx.x * blockDim.x + threadIdx.x;
    if (e >= Etot) return;
    int s, d;
    if (e < E) { s = src[e]; d = dst[e]; } else { s = d = e - E; }
    const float4* a4 = (const float4*)(asrc + (size_t)s * GAT_H);
    const float4* b4 = (const float4*)(adst + (size_t)d * GAT_H);
    float4 a0 = a4[0], a1 = a4[1], b0 = b4[0], b1 = b4[1];
    float p[8] = { a0.x+b0.x, a0.y+b0.y, a0.z+b0.z, a0.w+b0.w,
                   a1.x+b1.x, a1.y+b1.y, a1.z+b1.z, a1.w+b1.w };
    unsigned* mrow = mu + (size_t)d * GAT_H;
#pragma unroll
    for (int h = 0; h < 8; ++h) atomicMax(&mrow[h], fflip(lrelu(p[h])));
}

// =====================================================================
// K5: denom = segment_sum(exp(alpha - m))
// =====================================================================
__global__ __launch_bounds__(256) void edge_expsum(
    const int* __restrict__ src, const int* __restrict__ dst,
    const float* __restrict__ asrc, const float* __restrict__ adst,
    const unsigned* __restrict__ mu, float* __restrict__ denom,
    int E, int Etot)
{
    int e = blockIdx.x * blockDim.x + threadIdx.x;
    if (e >= Etot) return;
    int s, d;
    if (e < E) { s = src[e]; d = dst[e]; } else { s = d = e - E; }
    const float4* a4 = (const float4*)(asrc + (size_t)s * GAT_H);
    const float4* b4 = (const float4*)(adst + (size_t)d * GAT_H);
    float4 a0 = a4[0], a1 = a4[1], b0 = b4[0], b1 = b4[1];
    float p[8] = { a0.x+b0.x, a0.y+b0.y, a0.z+b0.z, a0.w+b0.w,
                   a1.x+b1.x, a1.y+b1.y, a1.z+b1.z, a1.w+b1.w };
    const unsigned* mrow = mu + (size_t)d * GAT_H;
    float* drow = denom + (size_t)d * GAT_H;
#pragma unroll
    for (int h = 0; h < 8; ++h)
        atomicAdd(&drow[h], __expf(lrelu(p[h]) - funflip(mrow[h])));
}

// =====================================================================
// K6: weighted message aggregation. One thread per (edge, head).
// =====================================================================
__global__ __launch_bounds__(256) void edge_aggregate(
    const int* __restrict__ src, const int* __restrict__ dst,
    const float* __restrict__ asrc, const float* __restrict__ adst,
    const unsigned* __restrict__ mu, const float* __restrict__ denom,
    const float* __restrict__ xp, float* __restrict__ agg,
    int E, int Etot)
{
    int idx = blockIdx.x * blockDim.x + threadIdx.x;
    if (idx >= Etot * GAT_H) return;
    int e = idx >> 3, h = idx & 7;
    int s, d;
    if (e < E) { s = src[e]; d = dst[e]; } else { s = d = e - E; }
    __builtin_prefetch(xp + (size_t)s * HD + h * GAT_D, 0, 1);  // global_prefetch
    float alpha = lrelu(asrc[(size_t)s * GAT_H + h] + adst[(size_t)d * GAT_H + h]);
    float m = funflip(mu[(size_t)d * GAT_H + h]);
    float w = __expf(alpha - m) / (denom[(size_t)d * GAT_H + h] + 1e-16f);
    const float4* xr = (const float4*)(xp + (size_t)s * HD + h * GAT_D);
    float4 v0 = xr[0], v1 = xr[1];
    float* ar = agg + (size_t)d * HD + h * GAT_D;
    atomicAdd(&ar[0], v0.x * w); atomicAdd(&ar[1], v0.y * w);
    atomicAdd(&ar[2], v0.z * w); atomicAdd(&ar[3], v0.w * w);
    atomicAdd(&ar[4], v1.x * w); atomicAdd(&ar[5], v1.y * w);
    atomicAdd(&ar[6], v1.z * w); atomicAdd(&ar[7], v1.w * w);
}

// =====================================================================
// K7: feat = relu(agg + b_gat)  (in place)
// =====================================================================
__global__ __launch_bounds__(256) void feat_relu(
    float* __restrict__ agg, const float* __restrict__ b_gat, int N)
{
    int i = blockIdx.x * blockDim.x + threadIdx.x;
    if (i >= N * HD) return;
    float v = agg[i] + b_gat[i & (HD - 1)];
    agg[i] = v > 0.f ? v : 0.f;
}

// =====================================================================
// K8: gi = feat @ W_ih^T + b_ih   [N,64] @ [64,48] -> [N,48] (bf16 WMMA)
// block = 192 thr (6 waves): 2 row tiles x 3 col tiles, 32 rows/block.
// Transposed [g][f] weight tile + padded A tile -> b128 fragment loads.
// =====================================================================
__global__ __launch_bounds__(192) void gi_gemm_wmma(
    const float* __restrict__ feat, const float* __restrict__ W_ih,
    const float* __restrict__ b_ih, float* __restrict__ gi)
{
    __shared__ __bf16 Wt2[G3 * 72];   // [g][f], stride 72 (144B, 16B-aligned)
    __shared__ __bf16 Al[32 * 40];    // [row][k], stride 40 (80B, 16B-aligned)

    const int tid  = threadIdx.x;
    const int lane = tid & 31;
    const int wave = tid >> 5;        // 0..5
    const int rt   = wave / 3;        // 0..1
    const int ct   = wave % 3;        // 0..2
    const int rowBase = blockIdx.x * 32;

    for (int i = tid; i < G3 * HD; i += 192) {
        int g = i >> 6, f = i & 63;
        Wt2[g * 72 + f] = (__bf16)W_ih[i];     // row-major copy, coalesced
    }

    v8f acc = {};
    const int lrow = lane & 15;
    const int kb   = (lane >> 4) * 8;

#pragma unroll
    for (int kk = 0; kk < HD; kk += 32) {
        __syncthreads();
        // stage A chunk: float4 load + packed 4x bf16 store (b64)
        for (int i = tid; i < 256; i += 192) {
            int r = i >> 3, c4 = i & 7;
            float4 f = *(const float4*)&feat[(size_t)(rowBase + r) * HD + kk + c4 * 4];
            v4bf o;
            o[0] = (__bf16)f.x; o[1] = (__bf16)f.y;
            o[2] = (__bf16)f.z; o[3] = (__bf16)f.w;
            *(v4bf*)&Al[r * 40 + c4 * 4] = o;
        }
        __syncthreads();

        const int arow = rt * 16 + lrow;
        v8bf alo = *(const v8bf*)&Al[arow * 40 + kb];
        v8bf ahi = *(const v8bf*)&Al[arow * 40 + kb + 16];
        v16bf a = __builtin_shufflevector(alo, ahi,
                                          0,1,2,3,4,5,6,7,8,9,10,11,12,13,14,15);
        const int bcol = ct * 16 + lrow;
        v8bf blo = *(const v8bf*)&Wt2[bcol * 72 + kk + kb];
        v8bf bhi = *(const v8bf*)&Wt2[bcol * 72 + kk + kb + 16];
        v16bf b = __builtin_shufflevector(blo, bhi,
                                          0,1,2,3,4,5,6,7,8,9,10,11,12,13,14,15);

        acc = __builtin_amdgcn_wmma_f32_16x16x32_bf16(
            false, a, false, b, (short)0, acc, false, false);
    }

    const int col  = ct * 16 + lrow;
    const int mofs = (lane >> 4) * 8;
    const float bias = b_ih[col];
#pragma unroll
    for (int r = 0; r < 8; ++r) {
        int row = rowBase + rt * 16 + r + mofs;
        gi[(size_t)row * G3 + col] = acc[r] + bias;
    }
}

// =====================================================================
// K9: GRU scan over T=2048 + linear head. One wave32 per batch element,
// 32 waves in one block. h replicated in registers, W_hh rows preloaded
// into per-lane registers -> the serial loop is pure VALU + shfl.
// =====================================================================
__global__ __launch_bounds__(1024) void gru_scan(
    const float* __restrict__ gi, const float* __restrict__ W_hh,
    const float* __restrict__ b_hh, const float* __restrict__ W_lin,
    const float* __restrict__ b_lin, float* __restrict__ out)
{
    const int tid  = threadIdx.x;
    const int b    = tid >> 5;        // batch element 0..31
    const int lane = tid & 31;
    const bool lo  = lane < 16;
    const int g0 = lane;              // lanes 0-15: r-gate, 16-31: z-gate
    const int g1 = 32 + (lane & 15);  // lanes 0-15: n-gate row

    float w0[HID], w1[HID];
#pragma unroll
    for (int j = 0; j < HID; ++j) w0[j] = W_hh[g0 * HID + j];
#pragma unroll
    for (int j = 0; j < HID; ++j) w1[j] = lo ? W_hh[g1 * HID + j] : 0.f;
    const float bh0 = b_hh[g0];
    const float bh1 = lo ? b_hh[g1] : 0.f;
    const float* gib = gi + (size_t)b * TSEQ * G3;

    float h[HID];
#pragma unroll
    for (int j = 0; j < HID; ++j) h[j] = 0.f;
    float myh = 0.f;                  // this lane's own h component (lane<16)

    for (int t = 0; t < TSEQ; ++t) {
        const float* grow = gib + t * G3;
        float gh0 = bh0, gh1 = bh1;
#pragma unroll
        for (int j = 0; j < HID; ++j) gh0 += h[j] * w0[j];
#pragma unroll
        for (int j = 0; j < HID; ++j) gh1 += h[j] * w1[j];
        float pre0 = grow[g0] + gh0;                  // r (lanes<16) / z (>=16)
        float s0 = 1.f / (1.f + __expf(-pre0));
        float z  = __shfl(s0, (lane + 16) & 31, 32);  // lane<16 pulls its z
        float hnew = 0.f;
        if (lo) {
            float r    = s0;
            float npre = grow[g1] + r * gh1;
            float nval = tanhf(npre);
            hnew = (1.f - z) * nval + z * myh;
            myh = hnew;
        }
#pragma unroll
        for (int j = 0; j < HID; ++j) h[j] = __shfl(hnew, j, 32);
    }

    // out[b] = h_last . W_lin + b_lin
    float p = lo ? myh * W_lin[lane] : 0.f;
#pragma unroll
    for (int off = 16; off > 0; off >>= 1) p += __shfl_xor(p, off, 32);
    if (lane == 0) out[b] = p + b_lin[0];
    if (lo) out[32 + b * HID + lane] = myh;    // h_last [1,B,HID]
}

// =====================================================================
// Launcher
// =====================================================================
extern "C" void kernel_launch(void* const* d_in, const int* in_sizes, int n_in,
                              void* d_out, int out_size, void* d_ws, size_t ws_size,
                              hipStream_t stream)
{
    const float* x       = (const float*)d_in[0];
    const int*   ei      = (const int*)d_in[1];   // [2,E] flat: src row, dst row
    const float* W_gat   = (const float*)d_in[3];
    const float* att_src = (const float*)d_in[4];
    const float* att_dst = (const float*)d_in[5];
    const float* b_gat   = (const float*)d_in[6];
    const float* W_ih    = (const float*)d_in[7];
    const float* W_hh    = (const float*)d_in[8];
    const float* b_ih    = (const float*)d_in[9];
    const float* b_hh    = (const float*)d_in[10];
    const float* W_lin   = (const float*)d_in[11];
    const float* b_lin   = (const float*)d_in[12];

    const int N = in_sizes[0] / FIN;      // 65536
    const int E = in_sizes[1] / 2;        // 1048576
    const int Etot = E + N;               // with self-loops

    // workspace layout (floats): ~55 MB total
    float*    xp    = (float*)d_ws;                 // N*64
    float*    asrc  = xp + (size_t)N * HD;          // N*8
    float*    adst  = asrc + (size_t)N * GAT_H;     // N*8
    unsigned* mu    = (unsigned*)(adst + (size_t)N * GAT_H);  // N*8
    float*    denom = (float*)mu + (size_t)N * GAT_H;         // N*8
    float*    agg   = denom + (size_t)N * GAT_H;    // N*64 (becomes feat)
    float*    gi    = agg + (size_t)N * HD;         // N*48

    const int* src = ei;
    const int* dst = ei + E;

    gat_proj_wmma<<<N / 32, 256, 0, stream>>>(x, W_gat, xp);
    att_logits<<<(N * GAT_H + 255) / 256, 256, 0, stream>>>(
        xp, att_src, att_dst, asrc, adst, N);
    init_scratch<<<(N * HD + 255) / 256, 256, 0, stream>>>(mu, denom, agg, N);
    edge_max<<<(Etot + 255) / 256, 256, 0, stream>>>(
        src, dst, asrc, adst, mu, E, Etot);
    edge_expsum<<<(Etot + 255) / 256, 256, 0, stream>>>(
        src, dst, asrc, adst, mu, denom, E, Etot);
    edge_aggregate<<<(Etot * GAT_H + 255) / 256, 256, 0, stream>>>(
        src, dst, asrc, adst, mu, denom, xp, agg, E, Etot);
    feat_relu<<<(N * HD + 255) / 256, 256, 0, stream>>>(agg, b_gat, N);
    gi_gemm_wmma<<<N / 32, 192, 0, stream>>>(agg, W_ih, b_ih, gi);
    gru_scan<<<1, 1024, 0, stream>>>(gi, W_hh, b_hh, W_lin, b_lin, (float*)d_out);
}
